// Block_21749714386969
// MI455X (gfx1250) — matrix operand order repaired and checked by reading, not compiled, not run on hardware
//
#include <hip/hip_runtime.h>
#include <cstdint>

// ---------------------------------------------------------------------------
// Problem constants (B=2, N=512, L=8192, C=256, H=8, D=32)
// ---------------------------------------------------------------------------
#define BB    2
#define NQ    512
#define LL    8192
#define CC    256
#define HH    8
#define DD    32
#define SCALE_F 0.17677669529663687f   // 1/sqrt(32)

typedef __attribute__((ext_vector_type(16))) _Float16 v16h;
typedef __attribute__((ext_vector_type(8)))  float    v8f;

union AFrag { uint32_t u[8]; v16h v; };

__device__ __forceinline__ v8f wmma_f16(const AFrag& a, const AFrag& b, v8f c) {
    // D = A(16x32 f16) x B(32x16 f16) + C(16x16 f32)
    return __builtin_amdgcn_wmma_f32_16x16x32_f16(
        /*neg_a=*/false, a.v, /*neg_b=*/false, b.v,
        /*c_mod=*/(short)0, c, /*reuse_a=*/false, /*reuse_b=*/false);
}

// ---------------------------------------------------------------------------
// fp32 -> f16 conversion (grid-stride)
// ---------------------------------------------------------------------------
__global__ void cvt_f32_f16(const float* __restrict__ in, _Float16* __restrict__ out, long n) {
    long i = (long)blockIdx.x * blockDim.x + threadIdx.x;
    long stride = (long)gridDim.x * blockDim.x;
    for (; i < n; i += stride) out[i] = (_Float16)in[i];
}

// ---------------------------------------------------------------------------
// Generic batched WMMA GEMM: C[m,n] = alpha * sum_k A[m,k]*Bt[n,k] (+bias, +act)
// A: (M,K) f16 row-major, Bt: (N,K) f16 row-major (i.e. B transposed).
// Batch offset = (z/inner)*so + (z%inner)*si for each of A/B/C.
// Register-blocked: one wave computes 32 x (16*NSUB) of C = 2 A-frags x NSUB
// B-frags, K-loop software-pipelined with double-buffered fragments.
// NSUB is a template constant and the wave id is readfirstlane'd, so all
// control flow is scalar: every WMMA executes with EXEC all-ones (ISA req).
// Requires M % 32 == 0, N % (16*NSUB) == 0.
// ---------------------------------------------------------------------------
template<int NSUB>
__global__ void __launch_bounds__(128) gemm_f16_wmma(
    const _Float16* __restrict__ A, long a_so, long a_si, int lda,
    const _Float16* __restrict__ Bt, long b_so, long b_si, int ldb,
    float* __restrict__ Cf, _Float16* __restrict__ Ch, long c_so, long c_si, int ldc,
    const float* __restrict__ bias, float alpha, int act,
    int M, int N, int K, int batch_inner)
{
    const int lane    = threadIdx.x;                                   // 0..31
    const int wave    = __builtin_amdgcn_readfirstlane(threadIdx.y);   // scalar 0..3
    const int tm_base = blockIdx.x * 32;
    const int tn_base = (blockIdx.y * 4 + wave) * (16 * NSUB);
    if (tn_base >= N) return;                                          // scalar branch

    const int z = blockIdx.z;
    const long aoff = (long)(z / batch_inner) * a_so + (long)(z % batch_inner) * a_si;
    const long boff = (long)(z / batch_inner) * b_so + (long)(z % batch_inner) * b_si;
    const long coff = (long)(z / batch_inner) * c_so + (long)(z % batch_inner) * c_si;

    const int m  = lane & 15;
    const int hb = lane >> 4;
    const _Float16* Ap0 = A + aoff + (long)(tm_base + m)      * lda;
    const _Float16* Ap1 = A + aoff + (long)(tm_base + 16 + m) * lda;
    const _Float16* Bp[NSUB];
#pragma unroll
    for (int j = 0; j < NSUB; ++j)
        Bp[j] = Bt + boff + (long)(tn_base + j * 16 + m) * ldb;

    v8f zero = {};
    v8f acc[2][NSUB];
#pragma unroll
    for (int i = 0; i < 2; ++i)
#pragma unroll
        for (int j = 0; j < NSUB; ++j) acc[i][j] = zero;

    AFrag a0[2], a1[2], b[2][NSUB];                 // double-buffered fragments

    auto load_stage = [&](int s, int k0) {
#pragma unroll
        for (int r = 0; r < 8; ++r) {
            const int ka = k0 + (hb * 16 + 2 * r);
            a0[s].u[r] = *(const uint32_t*)(Ap0 + ka);
            a1[s].u[r] = *(const uint32_t*)(Ap1 + ka);
        }
#pragma unroll
        for (int j = 0; j < NSUB; ++j)
#pragma unroll
            for (int r = 0; r < 8; ++r) {
                const int kb = k0 + (hb * 16 + 2 * r);
                b[s][j].u[r] = *(const uint32_t*)(Bp[j] + kb);
            }
    };
    auto do_wmma = [&](int s) {
#pragma unroll
        for (int j = 0; j < NSUB; ++j) {
            acc[0][j] = wmma_f16(a0[s], b[s][j], acc[0][j]);
            acc[1][j] = wmma_f16(a1[s], b[s][j], acc[1][j]);
        }
    };

    load_stage(0, 0);
    int k0 = 0;
    while (true) {                                   // unrolled-by-2 pipeline
        if (k0 + 32 < K) {
            load_stage(1, k0 + 32);
            __builtin_prefetch(Ap0 + k0 + 64, 0, 1); // gfx1250 global_prefetch_b8
            __builtin_prefetch(Bp[0] + k0 + 64, 0, 1);
        }
        do_wmma(0);
        k0 += 32;
        if (k0 >= K) break;
        if (k0 + 32 < K) load_stage(0, k0 + 32);
        do_wmma(1);
        k0 += 32;
        if (k0 >= K) break;
    }

#pragma unroll
    for (int j = 0; j < NSUB; ++j) {
        const int col = tn_base + j * 16 + m;
        const float bv = bias ? bias[col] : 0.0f;
#pragma unroll
        for (int i = 0; i < 2; ++i) {
#pragma unroll
            for (int r = 0; r < 8; ++r) {
                const int row = tm_base + i * 16 + r + hb * 8;
                float v = acc[i][j][r] * alpha + bv;
                if (act == 1) v = 0.5f * v * (1.0f + erff(v * 0.70710678118654752f));
                const long ci = coff + (long)row * ldc + col;
                if (Cf) Cf[ci] = v;
                if (Ch) Ch[ci] = (_Float16)v;
            }
        }
    }
}

// ---------------------------------------------------------------------------
// Transpose (B, Lt, ldin-slice) -> vT[((b*H+h)*D+d)*Lt + l], f16
// in element = in[(b*Lt + l)*ldin + h*D + d]
// ---------------------------------------------------------------------------
__global__ void transpose_dl(const _Float16* __restrict__ in, _Float16* __restrict__ out,
                             int Lt, int ldin) {
    long idx = (long)blockIdx.x * 256 + threadIdx.x;
    const long total = (long)BB * HH * DD * Lt;
    if (idx >= total) return;
    const int l = (int)(idx % Lt); long r = idx / Lt;
    const int d = (int)(r % DD);  r /= DD;
    const int h = (int)(r % HH);
    const int b = (int)(r / HH);
    out[idx] = in[((long)b * Lt + l) * ldin + h * DD + d];
}

// ---------------------------------------------------------------------------
// Row softmax over 512 columns, f32 in -> f16 out. One 256-thread block/row.
// ---------------------------------------------------------------------------
__global__ void __launch_bounds__(256) softmax512(const float* __restrict__ in,
                                                  _Float16* __restrict__ out) {
    const long row = blockIdx.x;
    const int  t   = threadIdx.x;
    const float* p = in + row * 512;
    __shared__ float red[256];
    float a = p[t], b = p[t + 256];
    red[t] = fmaxf(a, b); __syncthreads();
    for (int s = 128; s > 0; s >>= 1) { if (t < s) red[t] = fmaxf(red[t], red[t + s]); __syncthreads(); }
    const float mx = red[0]; __syncthreads();
    const float ea = __expf(a - mx), eb = __expf(b - mx);
    red[t] = ea + eb; __syncthreads();
    for (int s = 128; s > 0; s >>= 1) { if (t < s) red[t] += red[t + s]; __syncthreads(); }
    const float inv = 1.0f / red[0];
    out[row * 512 + t]       = (_Float16)(ea * inv);
    out[row * 512 + t + 256] = (_Float16)(eb * inv);
}

// ---------------------------------------------------------------------------
// LayerNorm( a + b ) over C=256, write f32 and/or f16. One block per row.
// ---------------------------------------------------------------------------
__global__ void __launch_bounds__(256) ln_residual(
    const float* __restrict__ a, const float* __restrict__ b,
    const float* __restrict__ g, const float* __restrict__ be,
    float* __restrict__ outF, _Float16* __restrict__ outH)
{
    const long row = blockIdx.x;
    const int  t   = threadIdx.x;
    const long i   = row * CC + t;
    const float x  = a[i] + b[i];
    __shared__ float red[256];
    red[t] = x; __syncthreads();
    for (int s = 128; s > 0; s >>= 1) { if (t < s) red[t] += red[t + s]; __syncthreads(); }
    const float mean = red[0] * (1.0f / CC); __syncthreads();
    const float d = x - mean;
    red[t] = d * d; __syncthreads();
    for (int s = 128; s > 0; s >>= 1) { if (t < s) red[t] += red[t + s]; __syncthreads(); }
    const float var = red[0] * (1.0f / CC);
    const float y = d * rsqrtf(var + 1e-5f) * g[t] + be[t];
    if (outF) outF[i] = y;
    if (outH) outH[i] = (_Float16)y;
}

// ---------------------------------------------------------------------------
// Fused cross-attention (flash, online softmax over L) + score-MLP mask.
// Block = 8 waves (one head each) x 32 lanes; grid = (N/16, B).
// Per 32-wide L tile: 2 score WMMAs/head -> LDS (all heads) -> mask MLP,
// p re-read from LDS in A-layout -> 2 p@v WMMAs/head.
// qc: (B,N,C) f16   kc: (B,L,C) f16   vT: (B,H,D,L) f16
// ---------------------------------------------------------------------------
__global__ void __launch_bounds__(256) ca_fused(
    const _Float16* __restrict__ qc, const _Float16* __restrict__ kc,
    const _Float16* __restrict__ vT,
    const float* __restrict__ l1w, const float* __restrict__ l1b,
    const float* __restrict__ l2w, const float* __restrict__ l2b,
    _Float16* __restrict__ ca_cat, float* __restrict__ mask)
{
    __shared__ __align__(16) float    s_lds[HH][16][32];
    __shared__ __align__(16) _Float16 p_lds[HH][16][32];
    __shared__ float wl1[64], bl1[8], wl2[8], bl2s;

    const int lane = threadIdx.x;
    const int h    = threadIdx.y;
    const int tid  = h * 32 + lane;
    if (tid < 64) wl1[tid] = l1w[tid];
    if (tid < 8)  { bl1[tid] = l1b[tid]; wl2[tid] = l2w[tid]; }
    if (tid == 0) bl2s = l2b[0];

    const int n0 = blockIdx.x * 16;
    const int b  = blockIdx.y;
    const int m  = lane & 15;
    const int hb = lane >> 4;

    // Q fragment (16 rows x K=32=D), loaded once
    AFrag qa;
    const _Float16* qp = qc + ((long)(b * NQ + n0 + m)) * CC + h * DD;
#pragma unroll
    for (int r = 0; r < 8; ++r) {
        const int ka = hb * 16 + 2 * r;
        qa.u[r] = *(const uint32_t*)(qp + ka);
    }

    v8f acc0 = {}, acc1 = {};
    float mrow[8], lrow[8];
#pragma unroll
    for (int r = 0; r < 8; ++r) { mrow[r] = -1e30f; lrow[r] = 0.0f; }

    const _Float16* kbase = kc + ((long)b * LL) * CC + h * DD;
    const _Float16* vbase = vT + ((long)(b * HH + h)) * DD * LL;

    __syncthreads();   // MLP weights visible

    for (int l0 = 0; l0 < LL; l0 += 32) {
        // --- scores: B fragments are k^T, lane holds column l ---
        AFrag kb0, kb1;
#pragma unroll
        for (int r = 0; r < 8; ++r) {
            const int kk = hb * 16 + 2 * r;                       // d index
            kb0.u[r] = *(const uint32_t*)(kbase + (long)(l0 + m)      * CC + kk);
            kb1.u[r] = *(const uint32_t*)(kbase + (long)(l0 + 16 + m) * CC + kk);
        }
        v8f s0 = {}, s1 = {};
        s0 = wmma_f16(qa, kb0, s0);
        s1 = wmma_f16(qa, kb1, s1);

        // --- online softmax update (row = r + 8*hb, lives in one 16-lane half) ---
#pragma unroll
        for (int r = 0; r < 8; ++r) {
            const float a = s0[r] * SCALE_F;
            const float c = s1[r] * SCALE_F;
            const int row = r + hb * 8;
            s_lds[h][row][m]      = a;
            s_lds[h][row][16 + m] = c;
            float t = fmaxf(a, c);
#pragma unroll
            for (int o = 1; o < 16; o <<= 1) t = fmaxf(t, __shfl_xor(t, o, 32));
            const float mn    = fmaxf(mrow[r], t);
            const float alpha = __expf(mrow[r] - mn);
            mrow[r] = mn;
            const float pa = __expf(a - mn), pc = __expf(c - mn);
            float rs = pa + pc;
#pragma unroll
            for (int o = 1; o < 16; o <<= 1) rs += __shfl_xor(rs, o, 32);
            lrow[r] = lrow[r] * alpha + rs;
            acc0[r] *= alpha; acc1[r] *= alpha;
            p_lds[h][row][m]      = (_Float16)pa;
            p_lds[h][row][16 + m] = (_Float16)pc;
        }
        __syncthreads();

        // --- mask MLP: 512 (n,l) positions, needs all 8 heads' scores ---
#pragma unroll
        for (int it = 0; it < 2; ++it) {
            const int pos = tid + it * 256;
            const int nn = pos >> 5, lc = pos & 31;
            float sh[8];
#pragma unroll
            for (int q2 = 0; q2 < 8; ++q2) sh[q2] = s_lds[q2][nn][lc];
            float mv = 0.0f;
#pragma unroll
            for (int j = 0; j < 8; ++j) {
                float f = bl1[j];
#pragma unroll
                for (int q2 = 0; q2 < 8; ++q2) f += wl1[j * 8 + q2] * sh[q2];
                mv += wl2[j] * fmaxf(f, 0.0f);
            }
            mask[((long)(b * NQ) + n0 + nn) * LL + l0 + lc] = fmaxf(mv + bl2s, 0.0f);
        }

        // --- p @ v : A frag from LDS (layout swap), B frags from vT ---
        AFrag pa2, vb0, vb1;
#pragma unroll
        for (int r = 0; r < 8; ++r) {
            const int ka = hb * 16 + 2 * r;  // l within tile
            pa2.u[r] = *(const uint32_t*)(&p_lds[h][m][ka]);
            const int kk = hb * 16 + 2 * r;                       // l within tile
            vb0.u[r] = *(const uint32_t*)(vbase + (long)m        * LL + l0 + kk);
            vb1.u[r] = *(const uint32_t*)(vbase + (long)(16 + m) * LL + l0 + kk);
        }
        acc0 = wmma_f16(pa2, vb0, acc0);
        acc1 = wmma_f16(pa2, vb1, acc1);
        __syncthreads();   // protect LDS before next tile
    }

    // --- epilogue: normalize, write (B,N,H*D) f16 ---
#pragma unroll
    for (int r = 0; r < 8; ++r) {
        const int row = r + hb * 8;
        const float inv = 1.0f / lrow[r];
        const long o = ((long)(b * NQ + n0 + row)) * CC + h * DD;
        ca_cat[o + m]      = (_Float16)(acc0[r] * inv);
        ca_cat[o + 16 + m] = (_Float16)(acc1[r] * inv);
    }
}

// ---------------------------------------------------------------------------
// Host orchestration
// ---------------------------------------------------------------------------
static inline char* ws_take(char*& cur, size_t bytes) {
    char* r = cur;
    cur += (bytes + 255) & ~(size_t)255;
    return r;
}

extern "C" void kernel_launch(void* const* d_in, const int* in_sizes, int n_in,
                              void* d_out, int out_size, void* d_ws, size_t ws_size,
                              hipStream_t stream) {
    const float* query     = (const float*)d_in[0];
    const float* key       = (const float*)d_in[1];
    const float* value     = (const float*)d_in[2];
    const float* sa_qkv_w  = (const float*)d_in[3];
    const float* sa_proj_w = (const float*)d_in[4];
    const float* sa_proj_b = (const float*)d_in[5];
    const float* norm3_g   = (const float*)d_in[6];
    const float* norm3_b   = (const float*)d_in[7];
    const float* q_w       = (const float*)d_in[8];
    const float* k_w       = (const float*)d_in[9];
    const float* v_w       = (const float*)d_in[10];
    const float* ca_proj_w = (const float*)d_in[11];
    const float* ca_proj_b = (const float*)d_in[12];
    const float* l1_w      = (const float*)d_in[13];
    const float* l1_b      = (const float*)d_in[14];
    const float* l2_w      = (const float*)d_in[15];
    const float* l2_b      = (const float*)d_in[16];
    const float* ln1_g     = (const float*)d_in[17];
    const float* ln1_b     = (const float*)d_in[18];
    const float* ln2_g     = (const float*)d_in[19];
    const float* ln2_b     = (const float*)d_in[20];
    const float* fc1_w     = (const float*)d_in[21];
    const float* fc1_b     = (const float*)d_in[22];
    const float* fc2_w     = (const float*)d_in[23];
    const float* fc2_b     = (const float*)d_in[24];

    float* x_out    = (float*)d_out;                    // (B,N,C)  = 262144
    float* mask_out = (float*)d_out + (long)BB*NQ*CC;   // (B,N,L)  = 8388608

    char* cur = (char*)d_ws;   // ~80 MB total
    const long nQ  = (long)BB*NQ*CC, nKV = (long)BB*LL*CC;
    _Float16* q16      = (_Float16*)ws_take(cur, nQ*2);
    _Float16* key16    = (_Float16*)ws_take(cur, nKV*2);
    _Float16* val16    = (_Float16*)ws_take(cur, nKV*2);
    _Float16* w_qkv16  = (_Float16*)ws_take(cur, 768L*256*2);
    _Float16* w_sap16  = (_Float16*)ws_take(cur, 256L*256*2);
    _Float16* w_q16    = (_Float16*)ws_take(cur, 256L*256*2);
    _Float16* w_k16    = (_Float16*)ws_take(cur, 256L*256*2);
    _Float16* w_v16    = (_Float16*)ws_take(cur, 256L*256*2);
    _Float16* w_cap16  = (_Float16*)ws_take(cur, 256L*256*2);
    _Float16* w_fc116  = (_Float16*)ws_take(cur, 1024L*256*2);
    _Float16* w_fc216  = (_Float16*)ws_take(cur, 256L*1024*2);
    _Float16* qkv16    = (_Float16*)ws_take(cur, (long)BB*NQ*768*2);
    _Float16* v_saT    = (_Float16*)ws_take(cur, (long)BB*HH*DD*NQ*2);
    float*    scores   = (float*)   ws_take(cur, (long)BB*HH*NQ*NQ*4);
    _Float16* probs    = (_Float16*)ws_take(cur, (long)BB*HH*NQ*NQ*2);
    _Float16* sa_cat   = (_Float16*)ws_take(cur, nQ*2);
    float*    sap_out  = (float*)   ws_take(cur, nQ*4);
    float*    x1_f32   = (float*)   ws_take(cur, nQ*4);
    _Float16* x1_f16   = (_Float16*)ws_take(cur, nQ*2);
    _Float16* qc16     = (_Float16*)ws_take(cur, nQ*2);
    _Float16* kc16     = (_Float16*)ws_take(cur, nKV*2);
    _Float16* vc16     = (_Float16*)ws_take(cur, nKV*2);
    _Float16* vcT      = (_Float16*)ws_take(cur, nKV*2);
    _Float16* ca_cat   = (_Float16*)ws_take(cur, nQ*2);
    float*    cap_out  = (float*)   ws_take(cur, nQ*4);
    float*    x2_f32   = (float*)   ws_take(cur, nQ*4);
    _Float16* x2_f16   = (_Float16*)ws_take(cur, nQ*2);
    _Float16* h16      = (_Float16*)ws_take(cur, (long)BB*NQ*1024*2);
    float*    fc2_out  = (float*)   ws_take(cur, nQ*4);
    (void)ws_size; (void)in_sizes; (void)n_in; (void)out_size;

    auto cvt = [&](const float* src, _Float16* dst, long n) {
        int blocks = (int)((n + 255) / 256); if (blocks > 4096) blocks = 4096;
        cvt_f32_f16<<<blocks, 256, 0, stream>>>(src, dst, n);
    };
    cvt(query, q16, nQ);           cvt(key, key16, nKV);        cvt(value, val16, nKV);
    cvt(sa_qkv_w, w_qkv16, 768L*256); cvt(sa_proj_w, w_sap16, 256L*256);
    cvt(q_w, w_q16, 256L*256);     cvt(k_w, w_k16, 256L*256);   cvt(v_w, w_v16, 256L*256);
    cvt(ca_proj_w, w_cap16, 256L*256);
    cvt(fc1_w, w_fc116, 1024L*256); cvt(fc2_w, w_fc216, 256L*1024);

    const dim3 gblk(32, 4, 1);
    auto gemm = [&](const _Float16* A, long aso, long asi, int lda,
                    const _Float16* Bt, long bso, long bsi, int ldb,
                    float* Cf, _Float16* Ch, long cso, long csi, int ldc,
                    const float* bias, float alpha, int act,
                    int M, int N, int K, int batch, int inner) {
        if (N % 64 == 0) {
            dim3 grid(M / 32, (N + 255) / 256, batch);     // wave = 32x64
            gemm_f16_wmma<4><<<grid, gblk, 0, stream>>>(A, aso, asi, lda, Bt, bso, bsi, ldb,
                                                        Cf, Ch, cso, csi, ldc,
                                                        bias, alpha, act, M, N, K, inner);
        } else {
            dim3 grid(M / 32, (N + 127) / 128, batch);     // wave = 32x32
            gemm_f16_wmma<2><<<grid, gblk, 0, stream>>>(A, aso, asi, lda, Bt, bso, bsi, ldb,
                                                        Cf, Ch, cso, csi, ldc,
                                                        bias, alpha, act, M, N, K, inner);
        }
    };

    // 1) fused QKV projection: (B*N,256) @ (768,256)^T -> f16 (B,N,768)
    gemm(q16,0,0,256, w_qkv16,0,0,256, nullptr,qkv16,0,0,768, nullptr,1.f,0, BB*NQ,768,256, 1,1);
    // 2) v_saT[b][h][d][n] from v slice of qkv
    transpose_dl<<<(BB*HH*DD*NQ)/256, 256, 0, stream>>>(qkv16 + 512, v_saT, NQ, 768);
    // 3) SA scores (b,h): q @ k^T * SCALE -> f32 (B,H,N,N)
    gemm(qkv16,(long)NQ*768,32,768, qkv16+256,(long)NQ*768,32,768,
         scores,nullptr,(long)HH*NQ*NQ,(long)NQ*NQ,NQ, nullptr,SCALE_F,0, NQ,NQ,DD, BB*HH,HH);
    // 4) softmax rows -> f16 probs
    softmax512<<<BB*HH*NQ, 256, 0, stream>>>(scores, probs);
    // 5) attn @ v -> sa_cat (B,N,C) f16  (C batch offset = b*N*C + h*D)
    gemm(probs,(long)HH*NQ*NQ,(long)NQ*NQ,NQ, v_saT,(long)HH*DD*NQ,(long)DD*NQ,NQ,
         nullptr,sa_cat,(long)NQ*CC,DD,CC, nullptr,1.f,0, NQ,DD,NQ, BB*HH,HH);
    // 6) sa_proj (+bias) -> f32
    gemm(sa_cat,0,0,256, w_sap16,0,0,256, sap_out,nullptr,0,0,256, sa_proj_b,1.f,0, BB*NQ,256,256, 1,1);
    // 7) x1 = LN(query + sa)
    ln_residual<<<BB*NQ, 256, 0, stream>>>(query, sap_out, norm3_g, norm3_b, x1_f32, x1_f16);
    // 8) CA projections
    gemm(x1_f16,0,0,256, w_q16,0,0,256, nullptr,qc16,0,0,256, nullptr,1.f,0, BB*NQ,256,256, 1,1);
    gemm(key16,0,0,256, w_k16,0,0,256, nullptr,kc16,0,0,256, nullptr,1.f,0, BB*LL,256,256, 1,1);
    gemm(val16,0,0,256, w_v16,0,0,256, nullptr,vc16,0,0,256, nullptr,1.f,0, BB*LL,256,256, 1,1);
    transpose_dl<<<(int)((long)BB*HH*DD*LL/256), 256, 0, stream>>>(vc16, vcT, LL, 256);
    // 9) fused flash cross-attention + mask MLP
    ca_fused<<<dim3(NQ/16, BB), dim3(32, 8), 0, stream>>>(
        qc16, kc16, vcT, l1_w, l1_b, l2_w, l2_b, ca_cat, mask_out);
    // 10) ca_proj (+bias) -> f32 ; x2 = LN(x1 + ca)
    gemm(ca_cat,0,0,256, w_cap16,0,0,256, cap_out,nullptr,0,0,256, ca_proj_b,1.f,0, BB*NQ,256,256, 1,1);
    ln_residual<<<BB*NQ, 256, 0, stream>>>(x1_f32, cap_out, ln1_g, ln1_b, x2_f32, x2_f16);
    // 11) FFN: fc1 (+bias, GELU) -> f16 ; fc2 (+bias) -> f32 ; final LN -> d_out
    gemm(x2_f16,0,0,256, w_fc116,0,0,256, nullptr,h16,0,0,1024, fc1_b,1.f,1, BB*NQ,1024,256, 1,1);
    gemm(h16,0,0,1024, w_fc216,0,0,1024, fc2_out,nullptr,0,0,256, fc2_b,1.f,0, BB*NQ,256,1024, 1,1);
    ln_residual<<<BB*NQ, 256, 0, stream>>>(x2_f32, fc2_out, ln2_g, ln2_b, x_out, nullptr);
}
